// GKATLayer_27556510171818
// MI455X (gfx1250) — matrix operand order repaired and checked by reading, not compiled, not run on hardware
//
#include <hip/hip_runtime.h>
#include <math.h>

// ---- types matching gfx1250 WMMA operand shapes (wave32) ----
typedef __attribute__((ext_vector_type(16))) __bf16 v16bf;  // A/B operand: 8 VGPRs
typedef __attribute__((ext_vector_type(8)))  __bf16 v8bf;   // 16-byte chunk
typedef __attribute__((ext_vector_type(8)))  float  v8f;    // C/D accum: 8 VGPRs

#define N_TOK   8192
#define IN_DIM  256
#define D_OUT   128
#define DE      144      // extended width: [ C@V (128) | C@1 (1) | zero pad (15) ]
#define NT      9        // DE/16 tiles
#define KSPLIT  8
#define JSPLIT  8

union ABu { v16bf v; v8bf h[2]; };

static __device__ __forceinline__ v8f wmma_bf16(v16bf a, v16bf b, v8f c) {
  // D = A(16x32) * B(32x16) + C, f32 accumulate
  return __builtin_amdgcn_wmma_f32_16x16x32_bf16(false, a, false, b, (short)0, c, false, false);
}

// ---------- prep kernels ----------
__global__ void k_cvt_feat(const float* __restrict__ f, __bf16* __restrict__ fb, int n) {
  int i = blockIdx.x * 256 + threadIdx.x;
  if (i < n) fb[i] = (__bf16)f[i];
}

__global__ void k_build_wt(const float* __restrict__ Wq, const float* __restrict__ Wk,
                           const float* __restrict__ Wv, __bf16* __restrict__ WT) {
  int i = blockIdx.x * 256 + threadIdx.x;   // over 384*256
  if (i >= 384 * IN_DIM) return;
  int n = i / IN_DIM, k = i % IN_DIM;       // WT[n][k] = W[k][n%128]
  const float* W = (n < 128) ? Wq : (n < 256) ? Wk : Wv;
  WT[i] = (__bf16)W[k * D_OUT + (n & 127)];
}

__global__ void k_vext_tail(__bf16* __restrict__ VextT) {
  int i = blockIdx.x * 256 + threadIdx.x;   // over 16*8192, rows 128..143
  if (i >= 16 * N_TOK) return;
  int r = i / N_TOK;
  VextT[(size_t)(128 + r) * N_TOK + (i % N_TOK)] = (r == 0) ? (__bf16)1.0f : (__bf16)0.0f;
}

__global__ void k_cvt_bf(const float* __restrict__ src, __bf16* __restrict__ dst, int n) {
  int i = blockIdx.x * 256 + threadIdx.x;
  if (i < n) dst[i] = (__bf16)src[i];
}

// ---------- QKV projection: [8192,256] @ [256,384] ----------
__global__ void __launch_bounds__(32)
k_proj(const __bf16* __restrict__ featb, const __bf16* __restrict__ WT,
       __bf16* __restrict__ Qb, __bf16* __restrict__ Kb, __bf16* __restrict__ VextT) {
  const int lane = threadIdx.x, l16 = lane & 15, half = lane >> 4;
  const int mb = blockIdx.x * 16;
  const int nb = blockIdx.y * 16;
  const __bf16* arow = featb + (size_t)(mb + l16) * IN_DIM;
  const __bf16* brow = WT    + (size_t)(nb + l16) * IN_DIM;
  v8f acc = {};
#pragma unroll
  for (int c = 0; c < IN_DIM / 32; ++c) {
    ABu a, b;
    a.h[0] = *(const v8bf*)(arow + c * 32 + 16 * half);
    a.h[1] = *(const v8bf*)(arow + c * 32 + 16 * half + 8);
    b.v    = *(const v16bf*)(brow + c * 32 + 16 * half);
    acc = wmma_bf16(a.v, b.v, acc);
  }
  const int ng = nb + l16;   // uniform branch per 16-wide tile
#pragma unroll
  for (int e = 0; e < 8; ++e) {
    int row = mb + e + 8 * half;
    __bf16 val = (__bf16)acc[e];
    if (ng < 128)      Qb[(size_t)row * D_OUT + ng] = val;
    else if (ng < 256) Kb[(size_t)row * D_OUT + (ng - 128)] = val;
    else               VextT[(size_t)(ng - 256) * N_TOK + row] = val;  // V transposed
  }
}

// ---------- CVext^T = (C @ [V|1|0])^T, streams C exactly once ----------
__global__ void __launch_bounds__(32)
k_cv(const float* __restrict__ C, const __bf16* __restrict__ VextT, float* __restrict__ CVTf) {
  const int lane = threadIdx.x, l16 = lane & 15, half = lane >> 4;
  const int kb = blockIdx.x * 16;                       // C row block
  const int jbase = blockIdx.y * (N_TOK / JSPLIT);      // j-split for occupancy
  v8f acc[NT];
#pragma unroll
  for (int t = 0; t < NT; ++t) acc[t] = (v8f){};
  const float* crow = C + (size_t)(kb + l16) * N_TOK;
  for (int j0 = jbase; j0 < jbase + N_TOK / JSPLIT; j0 += 32) {
    // pull the HBM stream ahead (global_prefetch_b8): ~4 iterations = 512B/row
    __builtin_prefetch(crow + j0 + 128, 0, 3);
    v8f c0 = *(const v8f*)(crow + j0 + 16 * half);
    v8f c1 = *(const v8f*)(crow + j0 + 16 * half + 8);
    ABu a;
#pragma unroll
    for (int i = 0; i < 8; ++i) { a.h[0][i] = (__bf16)c0[i]; a.h[1][i] = (__bf16)c1[i]; }
#pragma unroll
    for (int t = 0; t < NT; ++t) {
      v16bf b = *(const v16bf*)(VextT + (size_t)(16 * t + l16) * N_TOK + j0 + 16 * half);
      acc[t] = wmma_bf16(a.v, b, acc[t]);
    }
  }
#pragma unroll
  for (int t = 0; t < NT; ++t)
#pragma unroll
    for (int e = 0; e < 8; ++e)
      atomicAdd(CVTf + (size_t)(16 * t + l16) * N_TOK + kb + e + 8 * half, acc[t][e]);
}

// ---------- fused attention: O = exp(Q K^T * scale) @ CVext ----------
__global__ void __launch_bounds__(32)
k_flash(const __bf16* __restrict__ Qb, const __bf16* __restrict__ Kb,
        const __bf16* __restrict__ CVT, float* __restrict__ Of) {
  __shared__ __align__(64) __bf16 shP[16 * 32];   // one wave per block
  const int lane = threadIdx.x, l16 = lane & 15, half = lane >> 4;
  const int qb = blockIdx.x * 16;
  const int kbase = blockIdx.y * (N_TOK / KSPLIT);
  const float scale = 0.08838834764831845f;       // 1/sqrt(128)

  v16bf qa[4];                                    // Q block resident: 16x128
  {
    const __bf16* qrow = Qb + (size_t)(qb + l16) * D_OUT;
#pragma unroll
    for (int c = 0; c < 4; ++c) {
      ABu a;
      a.h[0] = *(const v8bf*)(qrow + c * 32 + 16 * half);
      a.h[1] = *(const v8bf*)(qrow + c * 32 + 16 * half + 8);
      qa[c] = a.v;
    }
  }
  v8f oacc[NT];
#pragma unroll
  for (int t = 0; t < NT; ++t) oacc[t] = (v8f){};

  for (int kb = kbase; kb < kbase + N_TOK / KSPLIT; kb += 32) {
    // prefetch next K block (L2/WGP$ resident stream)
    __builtin_prefetch(Kb + (size_t)(kb + 32 + lane) * D_OUT, 0, 3);
    // S = Q @ K^T for 32 K rows (two 16x16 tiles, k-dim 128 = 4 steps)
    v8f s0 = {}, s1 = {};
#pragma unroll
    for (int c = 0; c < 4; ++c) {
      v16bf b0 = *(const v16bf*)(Kb + (size_t)(kb + l16) * D_OUT + c * 32 + 16 * half);
      v16bf b1 = *(const v16bf*)(Kb + (size_t)(kb + 16 + l16) * D_OUT + c * 32 + 16 * half);
      s0 = wmma_bf16(qa[c], b0, s0);
      s1 = wmma_bf16(qa[c], b1, s1);
    }
    // P = exp(S*scale); C-layout -> LDS (row-major 16x32 bf16) -> A-layout
#pragma unroll
    for (int e = 0; e < 8; ++e) {
      int m = e + 8 * half;
      shP[m * 32 + l16]      = (__bf16)__expf(s0[e] * scale);
      shP[m * 32 + 16 + l16] = (__bf16)__expf(s1[e] * scale);
    }
    asm volatile("s_wait_dscnt 0" ::: "memory");   // same-wave RAW through LDS
    ABu pa;
    pa.h[0] = *(const v8bf*)(shP + l16 * 32 + 16 * half);
    pa.h[1] = *(const v8bf*)(shP + l16 * 32 + 16 * half + 8);
    // O += P @ CVext  (9 tiles: 128 CV cols + fused denominator col + pad)
#pragma unroll
    for (int t = 0; t < NT; ++t) {
      v16bf b = *(const v16bf*)(CVT + (size_t)(16 * t + l16) * N_TOK + kb + 16 * half);
      oacc[t] = wmma_bf16(pa.v, b, oacc[t]);
    }
  }
#pragma unroll
  for (int t = 0; t < NT; ++t)
#pragma unroll
    for (int e = 0; e < 8; ++e)
      atomicAdd(Of + (size_t)(qb + e + 8 * half) * DE + 16 * t + l16, oacc[t][e]);
}

// ---------- normalize + ELU ----------
__global__ void k_final(const float* __restrict__ Of, float* __restrict__ out) {
  int i = blockIdx.x * 256 + threadIdx.x;
  if (i >= N_TOK * D_OUT) return;
  int row = i / D_OUT, d = i % D_OUT;
  float num = Of[(size_t)row * DE + d];
  float den = Of[(size_t)row * DE + 128] + 1e-9f;
  float r = num / den;
  out[i] = (r > 0.f) ? r : (__expf(r) - 1.f);
}

extern "C" void kernel_launch(void* const* d_in, const int* in_sizes, int n_in,
                              void* d_out, int out_size, void* d_ws, size_t ws_size,
                              hipStream_t stream) {
  const float* feat = (const float*)d_in[0];
  // d_in[1] = bg (unused)
  const float* C    = (const float*)d_in[2];
  const float* Wq   = (const float*)d_in[3];
  const float* Wk   = (const float*)d_in[4];
  const float* Wv   = (const float*)d_in[5];
  float* out = (float*)d_out;

  char* ws = (char*)d_ws;
  size_t off = 0;
  auto alloc = [&](size_t bytes) -> void* {
    off = (off + 255) & ~(size_t)255;
    void* p = ws + off; off += bytes; return p;
  };
  __bf16* featb = (__bf16*)alloc((size_t)N_TOK * IN_DIM * 2);
  __bf16* WT    = (__bf16*)alloc((size_t)384 * IN_DIM * 2);
  __bf16* Qb    = (__bf16*)alloc((size_t)N_TOK * D_OUT * 2);
  __bf16* Kb    = (__bf16*)alloc((size_t)N_TOK * D_OUT * 2);
  __bf16* VextT = (__bf16*)alloc((size_t)DE * N_TOK * 2);
  float*  CVTf  = (float*) alloc((size_t)DE * N_TOK * 4);
  __bf16* CVT   = (__bf16*)alloc((size_t)DE * N_TOK * 2);
  float*  Of    = (float*) alloc((size_t)N_TOK * DE * 4);

  hipMemsetAsync(CVTf, 0, (size_t)DE * N_TOK * 4, stream);
  hipMemsetAsync(Of,   0, (size_t)N_TOK * DE * 4, stream);

  k_cvt_feat<<<(N_TOK * IN_DIM + 255) / 256, 256, 0, stream>>>(feat, featb, N_TOK * IN_DIM);
  k_build_wt<<<(384 * IN_DIM + 255) / 256, 256, 0, stream>>>(Wq, Wk, Wv, WT);
  k_vext_tail<<<(16 * N_TOK + 255) / 256, 256, 0, stream>>>(VextT);

  k_proj<<<dim3(N_TOK / 16, 384 / 16), 32, 0, stream>>>(featb, WT, Qb, Kb, VextT);
  k_cv<<<dim3(N_TOK / 16, JSPLIT), 32, 0, stream>>>(C, VextT, CVTf);
  k_cvt_bf<<<(DE * N_TOK + 255) / 256, 256, 0, stream>>>(CVTf, CVT, DE * N_TOK);
  k_flash<<<dim3(N_TOK / 16, KSPLIT), 32, 0, stream>>>(Qb, Kb, CVT, Of);
  k_final<<<(N_TOK * D_OUT + 255) / 256, 256, 0, stream>>>(Of, out);
}
